// PNAGNN_17961553232343
// MI455X (gfx1250) — matrix operand order repaired and checked by reading, not compiled, not run on hardware
//
#include <hip/hip_runtime.h>
#include <hip/hip_bf16.h>
#include <math.h>

typedef __attribute__((ext_vector_type(16))) _Float16 v16h;
typedef __attribute__((ext_vector_type(8)))  _Float16 v8h;
typedef __attribute__((ext_vector_type(8)))  float    v8f;

#define HDIM 64

// ---------- weight prep: [L][R][C] f32 -> [L][C][R] f16 (transposed) ----------
__global__ void wprep_kernel(const float* __restrict__ in, _Float16* __restrict__ out,
                             int Lc, int R, int C){
  long i = (long)blockIdx.x * blockDim.x + threadIdx.x;
  long total = (long)Lc * R * C;
  if (i >= total) return;
  int c = (int)(i % C);
  long t = i / C;
  int r = (int)(t % R);
  int l = (int)(t / R);
  out[((long)l * C + c) * R + r] = (_Float16)in[i];
}

__global__ void zeroi_kernel(int* p, long n){
  long i = (long)blockIdx.x * blockDim.x + threadIdx.x;
  if (i < n) p[i] = 0;
}

// ---------- embeddings ----------
__global__ void embed_atoms_kernel(const int* __restrict__ feat, const float* __restrict__ tab,
                                   float* __restrict__ x32, _Float16* __restrict__ x16, int N){
  long i = (long)blockIdx.x * blockDim.x + threadIdx.x;
  if (i >= (long)N * HDIM) return;
  int n = (int)(i >> 6), h = (int)(i & 63);
  float a = 0.0f;
#pragma unroll
  for (int c = 0; c < 9; ++c){
    int idx = feat[n * 9 + c];
    a += tab[((long)((c << 7) + idx)) * HDIM + h];   // ATOM_V = 128
  }
  x32[i] = a;
  x16[i] = (_Float16)a;
}

__global__ void embed_bonds_kernel(const int* __restrict__ feat, const float* __restrict__ tab,
                                   _Float16* __restrict__ e16, long Etot){
  long i = (long)blockIdx.x * blockDim.x + threadIdx.x;
  if (i >= Etot * HDIM) return;
  long e = i >> 6; int h = (int)(i & 63);
  float a = 0.0f;
#pragma unroll
  for (int c = 0; c < 3; ++c){
    int idx = feat[e * 3 + c];
    a += tab[((long)((c << 4) + idx)) * HDIM + h];   // BOND_V = 16
  }
  e16[i] = (_Float16)a;
}

// ---------- degrees (integer count, one-time) ----------
__global__ void degree_kernel(const int* __restrict__ eidx, int* __restrict__ degi, int E){
  long i = (long)blockIdx.x * blockDim.x + threadIdx.x;
  if (i >= E) return;
  atomicAdd(&degi[eidx[(long)E + i]], 1);
}

__global__ void degfin_kernel(const int* __restrict__ degi, float* __restrict__ invDc,
                              float* __restrict__ logD, float* __restrict__ invLogD, int N){
  long i = (long)blockIdx.x * blockDim.x + threadIdx.x;
  if (i >= N) return;
  float D = (float)degi[i];
  float dc = D > 1.0f ? D : 1.0f;
  invDc[i] = 1.0f / dc;
  float lg = logf(D + 1.0f);
  logD[i] = lg;
  invLogD[i] = 1.0f / lg;
}

// ---------- exclusive prefix scan over degrees (single block) ----------
__global__ __launch_bounds__(1024) void scan_kernel(const int* __restrict__ deg,
                                                    int* __restrict__ off, int N){
  __shared__ int buf[1024];
  __shared__ int carry;
  const int tid = threadIdx.x;
  if (tid == 0) carry = 0;
  __syncthreads();
  for (int base = 0; base < N; base += 1024){
    int v = (base + tid < N) ? deg[base + tid] : 0;
    buf[tid] = v;
    __syncthreads();
    for (int ofs = 1; ofs < 1024; ofs <<= 1){
      int t = (tid >= ofs) ? buf[tid - ofs] : 0;
      __syncthreads();
      buf[tid] += t;
      __syncthreads();
    }
    if (base + tid < N) off[base + tid] = carry + buf[tid] - v;
    __syncthreads();
    if (tid == 0) carry += buf[1023];
    __syncthreads();
  }
  if (tid == 0) off[N] = carry;
}

// ---------- counting-sort edge ids into CSR (one-time, E atomics) ----------
__global__ void csr_fill_kernel(const int* __restrict__ eidx, const int* __restrict__ off,
                                int* __restrict__ cursor, int* __restrict__ eids, int E){
  long i = (long)blockIdx.x * blockDim.x + threadIdx.x;
  if (i >= E) return;
  int d = eidx[(long)E + i];
  int pos = atomicAdd(&cursor[d], 1);
  eids[off[d] + pos] = (int)i;
}

// ---------- fused edge MLP -> m16[E,64] (f16), coalesced output ----------
// block = 256 threads = 8 waves; 128 edges/block, 16 edges per wave.
__global__ __launch_bounds__(256) void pretrans_kernel(
    const int* __restrict__ eidx, const _Float16* __restrict__ e16,
    const _Float16* __restrict__ x16,
    const _Float16* __restrict__ w1t, const float* __restrict__ b1,   // w1t: [64][192] f16
    const _Float16* __restrict__ w2t, const float* __restrict__ b2,   // w2t: [64][64] f16
    _Float16* __restrict__ m16, int E)
{
  __shared__ _Float16 zbuf[128 * 192];   // 48 KB: z rows, reused for hidden h / output m
  __shared__ int srcs[128];
  __shared__ int dsts[128];

  const int tid  = threadIdx.x;
  const int wave = tid >> 5;
  const int lane = tid & 31;
  const int mrow = lane & 15;
  const int khalf = lane >> 4;
  const int ebase = blockIdx.x * 128;

  for (int i = tid; i < 128; i += 256){
    int e = ebase + i; if (e >= E) e = E - 1;
    srcs[i] = eidx[e];
    dsts[i] = eidx[(long)E + e];
  }
  __syncthreads();

  // stage z = [x[dst] | x[src] | e] as 16-byte chunks (24 uint4 per 384B row)
  {
    const uint4* xu = (const uint4*)x16;   // 8 chunks per 64-f16 row
    const uint4* eu = (const uint4*)e16;
    uint4* zu = (uint4*)zbuf;              // 24 chunks per 192-f16 row
    for (int i = tid; i < 128 * 24; i += 256){
      int row = i / 24, c = i % 24;
      int e = ebase + row; if (e >= E) e = E - 1;
      uint4 v;
      if (c < 8)       v = xu[(long)dsts[row] * 8 + c];
      else if (c < 16) v = xu[(long)srcs[row] * 8 + (c - 8)];
      else             v = eu[(long)e * 8 + (c - 16)];
      zu[i] = v;
    }
  }
  __syncthreads();

  // GEMM1: [16 x 192] @ [192 x 64], K in 6 steps of 32
  const _Float16* zrow = &zbuf[(wave * 16 + mrow) * 192];
  v8f acc[4] = {v8f{}, v8f{}, v8f{}, v8f{}};
#pragma unroll
  for (int kb = 0; kb < 6; ++kb){
    v8h lo = *(const v8h*)(zrow + kb * 32 + khalf * 16);
    v8h hi = *(const v8h*)(zrow + kb * 32 + khalf * 16 + 8);
    v16h a = __builtin_shufflevector(lo, hi, 0,1,2,3,4,5,6,7,8,9,10,11,12,13,14,15);
#pragma unroll
    for (int t = 0; t < 4; ++t){
      v16h b = *(const v16h*)(w1t + (size_t)(t * 16 + mrow) * 192 + kb * 32 + khalf * 16);
      acc[t] = __builtin_amdgcn_wmma_f32_16x16x32_f16(false, a, false, b, (short)0, acc[t], false, false);
    }
  }

  // bias + relu -> f16 hidden, stored in this wave's own zbuf region (per-wave LDS in-order)
  _Float16* hrow = &zbuf[wave * 16 * 192];   // used as [16][64]
#pragma unroll
  for (int t = 0; t < 4; ++t){
    int col = t * 16 + mrow;
    float bv = b1[col];
#pragma unroll
    for (int r = 0; r < 8; ++r){
      int m = khalf * 8 + r;
      float v = acc[t][r] + bv;
      hrow[m * 64 + col] = (_Float16)(v > 0.0f ? v : 0.0f);
    }
  }

  // GEMM2: [16 x 64] @ [64 x 64]
  v8f acc2[4] = {v8f{}, v8f{}, v8f{}, v8f{}};
  const _Float16* zr2 = hrow + mrow * 64;
#pragma unroll
  for (int kb = 0; kb < 2; ++kb){
    v8h lo = *(const v8h*)(zr2 + kb * 32 + khalf * 16);
    v8h hi = *(const v8h*)(zr2 + kb * 32 + khalf * 16 + 8);
    v16h a = __builtin_shufflevector(lo, hi, 0,1,2,3,4,5,6,7,8,9,10,11,12,13,14,15);
#pragma unroll
    for (int t = 0; t < 4; ++t){
      v16h b = *(const v16h*)(w2t + (size_t)(t * 16 + mrow) * 64 + kb * 32 + khalf * 16);
      acc2[t] = __builtin_amdgcn_wmma_f32_16x16x32_f16(false, a, false, b, (short)0, acc2[t], false, false);
    }
  }

  // bias -> f16 m into this wave's LDS region (GEMM2 reads are complete, same-wave order)
#pragma unroll
  for (int t = 0; t < 4; ++t){
    int col = t * 16 + mrow;
    float bv = b2[col];
#pragma unroll
    for (int r = 0; r < 8; ++r){
      int m = khalf * 8 + r;
      hrow[m * 64 + col] = (_Float16)(acc2[t][r] + bv);
    }
  }

  // stream the wave's 16 contiguous edge rows (16x64 f16 = 2 KB) coalesced to global
  {
    const uint4* srcv = (const uint4*)hrow;                     // 128 x 16B
    uint4* dstv = (uint4*)(m16 + (size_t)(ebase + wave * 16) * HDIM);
    for (int i = lane; i < 128; i += 32){
      int row = i >> 3;                                         // 8 chunks per row
      if (ebase + wave * 16 + row < E) dstv[i] = srcv[i];
    }
  }
}

// ---------- CSR gather reduction: sum / sumsq / max / min (no atomics) ----------
// block 256 -> 4 nodes, 64 threads (one per feature) per node.
__global__ __launch_bounds__(256) void agg_kernel(
    const _Float16* __restrict__ m16, const int* __restrict__ off,
    const int* __restrict__ eids,
    float* __restrict__ sum, float* __restrict__ sumsq,
    float* __restrict__ mx, float* __restrict__ mn, int N)
{
  const int tid  = threadIdx.x;
  const int node = blockIdx.x * 4 + (tid >> 6);
  const int h    = tid & 63;
  if (node >= N) return;
  const int s0 = off[node], s1 = off[node + 1];
  float sm = 0.0f, sq = 0.0f;
  float ma = -3.402823466e38f, mi = 3.402823466e38f;
  for (int i = s0; i < s1; ++i){
    if (i + 4 < s1)                                  // hide gather latency
      __builtin_prefetch(m16 + (long)eids[i + 4] * HDIM + h, 0, 0);
    int e = eids[i];
    float v = (float)m16[(long)e * HDIM + h];        // 64 lanes cover one 128B row
    sm += v; sq += v * v;
    ma = v > ma ? v : ma;
    mi = v < mi ? v : mi;
  }
  if (s1 == s0){ ma = 0.0f; mi = 0.0f; }             // mimic isfinite -> 0 (deg==0)
  long o = (long)node * HDIM + h;
  sum[o] = sm; sumsq[o] = sq; mx[o] = ma; mn[o] = mi;
}

// ---------- h_cat chunk synthesis (8 consecutive f16 of the 832-wide row) ----------
struct NodeStats {
  const _Float16* x16; const float* sum; const float* sumsq;
  const float* mx; const float* mn;
  const float* invDc; const float* logD; const float* invLogD;
};

__device__ __forceinline__ v8h hcat_chunk(const NodeStats& S, int node, int k0){
  int b = k0 >> 6;        // 64-wide block: 0 = x, 1..12 = {mean,max,min,std} x {1,logD,1/logD}
  int o = k0 & 63;
  long base = (long)node * HDIM + o;
  if (b == 0) return *(const v8h*)(S.x16 + base);
  int sb = (b - 1) / 3;
  int mu = (b - 1) % 3;
  float mval = (mu == 0) ? 1.0f : (mu == 1 ? S.logD[node] : S.invLogD[node]);
  float id = S.invDc[node];
  v8h out;
#pragma unroll
  for (int j = 0; j < 8; ++j){
    float v;
    if (sb == 0)      v = S.sum[base + j] * id;
    else if (sb == 1) v = S.mx[base + j];
    else if (sb == 2) v = S.mn[base + j];
    else {
      float me = S.sum[base + j] * id;
      float ms = S.sumsq[base + j] * id;
      float var = ms - me * me;
      var = var > 0.0f ? var : 0.0f;
      v = sqrtf(var + 1e-5f);
    }
    out[j] = (_Float16)(v * mval);
  }
  return out;
}

// ---------- node MLP + residual ----------
// block = 256 = 8 waves; 128 nodes/block, 16 nodes per wave.
__global__ __launch_bounds__(256) void posttrans_kernel(
    const _Float16* __restrict__ x16, const float* __restrict__ x32old,
    float* __restrict__ xout, _Float16* __restrict__ x16out, int writeF16,
    const float* __restrict__ sum, const float* __restrict__ sumsq,
    const float* __restrict__ mx, const float* __restrict__ mn,
    const float* __restrict__ invDc, const float* __restrict__ logD,
    const float* __restrict__ invLogD,
    const _Float16* __restrict__ w1t, const float* __restrict__ b1,   // w1t: [64][832] f16
    const _Float16* __restrict__ w2t, const float* __restrict__ b2,   // w2t: [64][64] f16
    int N)
{
  __shared__ _Float16 hbuf[8 * 16 * 64];   // 16 KB

  const int tid  = threadIdx.x;
  const int wave = tid >> 5;
  const int lane = tid & 31;
  const int mrow = lane & 15;
  const int khalf = lane >> 4;
  const int nbase = blockIdx.x * 128 + wave * 16;

  int nodeA = nbase + mrow; if (nodeA >= N) nodeA = N - 1;
  NodeStats S{ x16, sum, sumsq, mx, mn, invDc, logD, invLogD };

  // GEMM1: [16 x 832] @ [832 x 64], K in 26 steps of 32; A synthesized on the fly
  v8f acc[4] = {v8f{}, v8f{}, v8f{}, v8f{}};
  for (int kb = 0; kb < 26; ++kb){
    v8h lo = hcat_chunk(S, nodeA, kb * 32 + khalf * 16);
    v8h hi = hcat_chunk(S, nodeA, kb * 32 + khalf * 16 + 8);
    v16h a = __builtin_shufflevector(lo, hi, 0,1,2,3,4,5,6,7,8,9,10,11,12,13,14,15);
#pragma unroll
    for (int t = 0; t < 4; ++t){
      v16h b = *(const v16h*)(w1t + (size_t)(t * 16 + mrow) * 832 + kb * 32 + khalf * 16);
      acc[t] = __builtin_amdgcn_wmma_f32_16x16x32_f16(false, a, false, b, (short)0, acc[t], false, false);
    }
  }

  _Float16* hrow = &hbuf[wave * 1024];     // [16][64]
#pragma unroll
  for (int t = 0; t < 4; ++t){
    int col = t * 16 + mrow;
    float bv = b1[col];
#pragma unroll
    for (int r = 0; r < 8; ++r){
      int m = khalf * 8 + r;
      float v = acc[t][r] + bv;
      hrow[m * 64 + col] = (_Float16)(v > 0.0f ? v : 0.0f);
    }
  }

  // GEMM2: [16 x 64] @ [64 x 64]
  v8f acc2[4] = {v8f{}, v8f{}, v8f{}, v8f{}};
  const _Float16* zr2 = hrow + mrow * 64;
#pragma unroll
  for (int kb = 0; kb < 2; ++kb){
    v8h lo = *(const v8h*)(zr2 + kb * 32 + khalf * 16);
    v8h hi = *(const v8h*)(zr2 + kb * 32 + khalf * 16 + 8);
    v16h a = __builtin_shufflevector(lo, hi, 0,1,2,3,4,5,6,7,8,9,10,11,12,13,14,15);
#pragma unroll
    for (int t = 0; t < 4; ++t){
      v16h b = *(const v16h*)(w2t + (size_t)(t * 16 + mrow) * 64 + kb * 32 + khalf * 16);
      acc2[t] = __builtin_amdgcn_wmma_f32_16x16x32_f16(false, a, false, b, (short)0, acc2[t], false, false);
    }
  }

  // bias + residual; in-place node update is safe (each node owned by one wave)
#pragma unroll
  for (int t = 0; t < 4; ++t){
    int col = t * 16 + mrow;
    float bv = b2[col];
#pragma unroll
    for (int r = 0; r < 8; ++r){
      int m = khalf * 8 + r;
      int node = nbase + m;
      if (node < N){
        long o = (long)node * HDIM + col;
        float v = acc2[t][r] + bv + x32old[o];
        xout[o] = v;
        if (writeF16) x16out[o] = (_Float16)v;
      }
    }
  }
}

// ======================================================================
extern "C" void kernel_launch(void* const* d_in, const int* in_sizes, int n_in,
                              void* d_out, int out_size, void* d_ws, size_t ws_size,
                              hipStream_t stream) {
  const int*   x_feat   = (const int*)  d_in[0];
  const int*   eidx     = (const int*)  d_in[1];
  const int*   eattr    = (const int*)  d_in[2];
  const float* atom_emb = (const float*)d_in[3];
  const float* bond_emb = (const float*)d_in[4];
  const float* pre_w1   = (const float*)d_in[5];
  const float* pre_b1   = (const float*)d_in[6];
  const float* pre_w2   = (const float*)d_in[7];
  const float* pre_b2   = (const float*)d_in[8];
  const float* post_w1  = (const float*)d_in[9];
  const float* post_b1  = (const float*)d_in[10];
  const float* post_w2  = (const float*)d_in[11];
  const float* post_b2  = (const float*)d_in[12];

  const int N  = in_sizes[0] / 9;
  const int E  = in_sizes[2] / 3;
  const int Lc = in_sizes[6] / HDIM;          // pre_b1: [L, 64]
  float* out = (float*)d_out;

  // ---- carve workspace ----
  char* p = (char*)d_ws;
  auto carve = [&](size_t bytes) -> char* {
    char* r = p;
    p += (bytes + 255) & ~(size_t)255;
    return r;
  };
  _Float16* e16      = (_Float16*)carve((size_t)E * HDIM * 2);
  _Float16* m16      = (_Float16*)carve((size_t)E * HDIM * 2);
  _Float16* x16      = (_Float16*)carve((size_t)N * HDIM * 2);
  float*    x32      = (float*)   carve((size_t)N * HDIM * 4);
  float*    aggsum   = (float*)   carve((size_t)N * HDIM * 4);
  float*    aggsq    = (float*)   carve((size_t)N * HDIM * 4);
  float*    aggmax   = (float*)   carve((size_t)N * HDIM * 4);
  float*    aggmin   = (float*)   carve((size_t)N * HDIM * 4);
  int*      degi     = (int*)     carve((size_t)N * 4);
  int*      csr_off  = (int*)     carve((size_t)(N + 1) * 4);
  int*      cursor   = (int*)     carve((size_t)N * 4);
  int*      csr_eid  = (int*)     carve((size_t)E * 4);
  float*    invDc    = (float*)   carve((size_t)N * 4);
  float*    logD     = (float*)   carve((size_t)N * 4);
  float*    invLogD  = (float*)   carve((size_t)N * 4);
  _Float16* pre_w1t  = (_Float16*)carve((size_t)Lc * 64 * 192 * 2);
  _Float16* pre_w2t  = (_Float16*)carve((size_t)Lc * 64 * 64  * 2);
  _Float16* post_w1t = (_Float16*)carve((size_t)Lc * 64 * 832 * 2);
  _Float16* post_w2t = (_Float16*)carve((size_t)Lc * 64 * 64  * 2);

  auto blks = [](long n, int b){ return (int)((n + b - 1) / b); };

  // ---- weight prep (transpose + f16) ----
  wprep_kernel<<<blks((long)Lc*192*64, 256), 256, 0, stream>>>(pre_w1,  pre_w1t,  Lc, 192, 64);
  wprep_kernel<<<blks((long)Lc*64*64,  256), 256, 0, stream>>>(pre_w2,  pre_w2t,  Lc, 64,  64);
  wprep_kernel<<<blks((long)Lc*832*64, 256), 256, 0, stream>>>(post_w1, post_w1t, Lc, 832, 64);
  wprep_kernel<<<blks((long)Lc*64*64,  256), 256, 0, stream>>>(post_w2, post_w2t, Lc, 64,  64);

  // ---- embeddings ----
  embed_atoms_kernel<<<blks((long)N*HDIM, 256), 256, 0, stream>>>(x_feat, atom_emb, x32, x16, N);
  embed_bonds_kernel<<<blks((long)E*HDIM, 256), 256, 0, stream>>>(eattr, bond_emb, e16, (long)E);

  // ---- degrees + CSR (one-time) ----
  zeroi_kernel<<<blks(N, 256), 256, 0, stream>>>(degi, N);
  degree_kernel<<<blks(E, 256), 256, 0, stream>>>(eidx, degi, E);
  degfin_kernel<<<blks(N, 256), 256, 0, stream>>>(degi, invDc, logD, invLogD, N);
  scan_kernel<<<1, 1024, 0, stream>>>(degi, csr_off, N);
  zeroi_kernel<<<blks(N, 256), 256, 0, stream>>>(cursor, N);
  csr_fill_kernel<<<blks(E, 256), 256, 0, stream>>>(eidx, csr_off, cursor, csr_eid, E);

  // ---- layers ----
  for (int l = 0; l < Lc; ++l){
    pretrans_kernel<<<blks(E, 128), 256, 0, stream>>>(
        eidx, e16, x16,
        pre_w1t + (size_t)l*64*192, pre_b1 + l*HDIM,
        pre_w2t + (size_t)l*64*64,  pre_b2 + l*HDIM,
        m16, E);

    agg_kernel<<<blks(N, 4), 256, 0, stream>>>(
        m16, csr_off, csr_eid, aggsum, aggsq, aggmax, aggmin, N);

    const bool last = (l == Lc - 1);
    posttrans_kernel<<<blks(N, 128), 256, 0, stream>>>(
        x16, x32,
        last ? out : x32, x16, last ? 0 : 1,
        aggsum, aggsq, aggmax, aggmin,
        invDc, logD, invLogD,
        post_w1t + (size_t)l*64*832, post_b1 + l*HDIM,
        post_w2t + (size_t)l*64*64,  post_b2 + l*HDIM,
        N);
  }
}
